// Attention_42880953483401
// MI455X (gfx1250) — matrix operand order repaired and checked by reading, not compiled, not run on hardware
//
#include <hip/hip_runtime.h>
#include <hip/hip_bf16.h>

// ---------------------------------------------------------------------------
// Attention pipeline for MI455X (gfx1250, wave32, WMMA).
//   B=8, SQ=SK=2048, DQ=DK=DV=1024
//   Pass 1: q/k/v projections (fp32 in -> bf16 out, bias add, v transposed)
//   Pass 2: S = (k . q^T) * scale + mask^T * (-1e9)   (fp32 out)
//   Pass 3: row softmax over SQ -> attn fp32 (output) + bf16 (workspace)
//   Pass 4: context = attn_bf16 . v^T                 (fp32 out)
// All GEMMs: C[M,N] = A[M,K] * B[N,K]^T with K contiguous in both operands,
// computed with v_wmma_f32_16x16x32_bf16. The all-bf16 GEMMs (pass 2/4) stage
// tiles with GLOBAL_LOAD_ASYNC_TO_LDS_B128 + double buffering (ASYNCcnt).
// ---------------------------------------------------------------------------

typedef __attribute__((ext_vector_type(16))) __bf16 v16bf;
typedef __attribute__((ext_vector_type(8)))  float  v8f;
typedef __attribute__((ext_vector_type(4)))  int    i4x;
typedef __attribute__((ext_vector_type(4)))  float  fx4;

#define TILE_LDS_STRIDE 40   // halfs per row: 80B, 20 banks -> conflict-free
#define TILE_HALFS      (128 * TILE_LDS_STRIDE)

union FragBF {
    i4x   i2[2];
    v16bf v;
};

union Pack8BF {
    __bf16 h[8];
    i4x    v;
};

// Synchronous staging of a 128x32 tile into LDS with fp32 -> bf16 conversion.
__device__ inline void load_tile_f32(const float* g, int ld, int tile_row, int k0,
                                     unsigned short* lds, int tid)
{
    #pragma unroll
    for (int s = 0; s < 2; ++s) {
        int id  = tid + s * 256;        // 512 units of 8 elements
        int row = id >> 2;              // 4 units per 32-wide row
        int hc  = id & 3;
        const fx4* gp = (const fx4*)(g + (size_t)(tile_row + row) * ld + k0 + hc * 8);
        fx4 a = gp[0];
        fx4 b = gp[1];
        Pack8BF u;
        #pragma unroll
        for (int j = 0; j < 4; ++j) {
            u.h[j]     = (__bf16)a[j];
            u.h[4 + j] = (__bf16)b[j];
        }
        *(i4x*)(&lds[row * TILE_LDS_STRIDE + hc * 8]) = u.v;
    }
}

// Async staging of a 128x32 bf16 tile: one global_load_async_to_lds_b128 per
// 16-byte chunk. ldsOff is the 32-bit DS offset of the tile buffer base.
__device__ inline void async_tile_bf16(const __bf16* gbase, int ld, int tile_row,
                                       int k0, unsigned ldsOff, int tid)
{
    #pragma unroll
    for (int s = 0; s < 2; ++s) {
        int id  = tid + s * 256;
        int row = id >> 2;
        int kc  = id & 3;
        unsigned voff = ((unsigned)(tile_row + row) * (unsigned)ld
                         + (unsigned)(k0 + kc * 8)) * 2u;          // byte offset
        unsigned dso  = ldsOff + (unsigned)(row * TILE_LDS_STRIDE + kc * 8) * 2u;
        asm volatile("global_load_async_to_lds_b128 %0, %1, %2"
                     :: "v"(dso), "v"(voff), "s"(gbase)
                     : "memory");
    }
}

__device__ inline void wait_asynccnt0()
{
    asm volatile("s_wait_asynccnt 0x0" ::: "memory");
}

// Load fragments from an LDS tile pair and accumulate 2x4 WMMA tiles.
__device__ inline void mma_step(const unsigned short* pA, const unsigned short* pB,
                                int aRow, int ka, int bCol, int kb, v8f (&acc)[2][4])
{
    FragBF fa[2];
    #pragma unroll
    for (int mt = 0; mt < 2; ++mt) {
        const int r = aRow + mt * 16;
        fa[mt].i2[0] = *(const i4x*)(&pA[r * TILE_LDS_STRIDE + ka]);
        fa[mt].i2[1] = *(const i4x*)(&pA[r * TILE_LDS_STRIDE + ka + 16]);
    }
    FragBF fb[4];
    #pragma unroll
    for (int nt = 0; nt < 4; ++nt) {
        const int c = bCol + nt * 16;
        fb[nt].i2[0] = *(const i4x*)(&pB[c * TILE_LDS_STRIDE + kb]);
        fb[nt].i2[1] = *(const i4x*)(&pB[c * TILE_LDS_STRIDE + kb + 16]);
    }
    #pragma unroll
    for (int mt = 0; mt < 2; ++mt)
        #pragma unroll
        for (int nt = 0; nt < 4; ++nt)
            acc[mt][nt] = __builtin_amdgcn_wmma_f32_16x16x32_bf16(
                false, fa[mt].v, false, fb[nt].v,
                (short)0, acc[mt][nt], false, false);
}

// MODE 0: bf16 out, row-major, + bias            (projection q/k)
// MODE 1: bf16 out, transposed per batch, + bias (projection v -> v^T)
// MODE 2: fp32 out, *scale + mask^T * (-1e9)     (scores)
// MODE 3: fp32 out                               (context)
template<int MODE, bool AF32, bool BF32>
__global__ __launch_bounds__(256)
void gemm_wmma_k(const void* __restrict__ Ag, const void* __restrict__ Bg,
                 void* __restrict__ Cg,
                 const float* __restrict__ bias, const int* __restrict__ mask,
                 int M, int N, int K,
                 long long sA, long long sB, long long sC, long long sM,
                 int SQc, float scale)
{
    constexpr bool ASYNC = (!AF32) && (!BF32);

    __shared__ __align__(16) unsigned short lA[2][TILE_HALFS];
    __shared__ __align__(16) unsigned short lB[2][TILE_HALFS];

    const int tid    = threadIdx.x;
    const int lane   = tid & 31;
    const int wid    = tid >> 5;
    const int wave_m = wid >> 1;     // 0..3 -> 32-row strip
    const int wave_n = wid & 1;      // 0..1 -> 64-col strip
    const int tile_m = blockIdx.y * 128;
    const int tile_n = blockIdx.x * 128;
    const int bz     = blockIdx.z;

    const char* Abase = (const char*)Ag + (size_t)bz * (size_t)sA * (AF32 ? 4 : 2);
    const char* Bbase = (const char*)Bg + (size_t)bz * (size_t)sB * (BF32 ? 4 : 2);

    v8f acc[2][4];
    #pragma unroll
    for (int mt = 0; mt < 2; ++mt)
        #pragma unroll
        for (int nt = 0; nt < 4; ++nt)
            #pragma unroll
            for (int e = 0; e < 8; ++e)
                acc[mt][nt][e] = 0.0f;

    const int aRow = (lane & 15) + wave_m * 32;   // fragment row (ISA A layout)
    const int ka   = (lane >> 4) * 8;             // K base: lanes 0-15 -> 0, 16-31 -> 8
    const int bCol = (lane & 15) + wave_n * 64;   // fragment col (ISA B layout)
    const int kb   = (lane >> 4) * 8;             // K base: lanes 0-15 -> 0, 16-31 -> 8

    if (ASYNC) {
        // bf16 operands: async LDS staging + double buffering.
        const __bf16* Abf = (const __bf16*)Abase;
        const __bf16* Bbf = (const __bf16*)Bbase;
        const unsigned la0 = (unsigned)(size_t)&lA[0][0];
        const unsigned lb0 = (unsigned)(size_t)&lB[0][0];
        const unsigned bufBytes = (unsigned)TILE_HALFS * 2u;
        const int nk = K >> 5;

        async_tile_bf16(Abf, K, tile_m, 0, la0, tid);
        async_tile_bf16(Bbf, K, tile_n, 0, lb0, tid);

        for (int kt = 0; kt < nk; ++kt) {
            const int cur = kt & 1;
            wait_asynccnt0();     // my async fills landed in LDS
            __syncthreads();      // everyone's fills landed; prev buffer fully read
            if (kt + 1 < nk) {
                const int nxt = cur ^ 1;
                async_tile_bf16(Abf, K, tile_m, (kt + 1) * 32,
                                la0 + (unsigned)nxt * bufBytes, tid);
                async_tile_bf16(Bbf, K, tile_n, (kt + 1) * 32,
                                lb0 + (unsigned)nxt * bufBytes, tid);
            }
            mma_step(&lA[cur][0], &lB[cur][0], aRow, ka, bCol, kb, acc);
        }
    } else {
        // fp32 operands: synchronous load + convert into single buffer.
        for (int k0 = 0; k0 < K; k0 += 32) {
            __syncthreads();
            load_tile_f32((const float*)Abase, K, tile_m, k0, &lA[0][0], tid);
            load_tile_f32((const float*)Bbase, K, tile_n, k0, &lB[0][0], tid);
            __syncthreads();
            mma_step(&lA[0][0], &lB[0][0], aRow, ka, bCol, kb, acc);
        }
    }

    // Epilogue. C fragment layout: lanes 0-15 col=lane, rows = vgpr; lanes
    // 16-31 col=lane-16, rows = vgpr+8.
    #pragma unroll
    for (int mt = 0; mt < 2; ++mt) {
        const int rowBase = tile_m + wave_m * 32 + mt * 16 + ((lane >> 4) * 8);
        #pragma unroll
        for (int nt = 0; nt < 4; ++nt) {
            const int col = tile_n + wave_n * 64 + nt * 16 + (lane & 15);
            v8f a = acc[mt][nt];
            if (MODE == 0) {
                __bf16* Co = (__bf16*)Cg;
                const float bv = bias[col];
                #pragma unroll
                for (int i = 0; i < 8; ++i)
                    Co[(size_t)(rowBase + i) * N + col] = (__bf16)(a[i] + bv);
            } else if (MODE == 1) {
                __bf16* Co = (__bf16*)Cg;
                const float bv = bias[col];
                #pragma unroll
                for (int i = 0; i < 8; ++i) {
                    const int rg = rowBase + i;
                    const int bb = rg / SQc;
                    const int q  = rg - bb * SQc;
                    Co[((size_t)bb * N + col) * SQc + q] = (__bf16)(a[i] + bv);
                }
            } else if (MODE == 2) {
                float* Co = (float*)Cg + (size_t)bz * sC;
                const int* mp = mask + (size_t)bz * sM + (size_t)col * SQc + rowBase;
                union { i4x v; int s[4]; } m0, m1;
                m0.v = *(const i4x*)(mp);
                m1.v = *(const i4x*)(mp + 4);
                #pragma unroll
                for (int i = 0; i < 8; ++i) {
                    const int mv = (i < 4) ? m0.s[i & 3] : m1.s[i & 3];
                    Co[(size_t)(rowBase + i) * N + col] =
                        a[i] * scale + (float)mv * (-1.0e9f);
                }
            } else {
                float* Co = (float*)Cg + (size_t)bz * sC;
                #pragma unroll
                for (int i = 0; i < 8; ++i)
                    Co[(size_t)(rowBase + i) * N + col] = a[i];
            }
        }
    }
}

// Row softmax over L=2048 fp32; writes fp32 (output) and bf16 (workspace).
__global__ __launch_bounds__(256)
void softmax_rows(const float* __restrict__ S, float* __restrict__ outf,
                  void* __restrict__ outbf, int L)
{
    __shared__ float red[256];
    const int row = blockIdx.x;
    const int t   = threadIdx.x;
    const float* p = S + (size_t)row * L;

    float loc[8];
    float mx = -3.4e38f;
    #pragma unroll
    for (int i = 0; i < 8; ++i) {
        loc[i] = p[t + i * 256];
        mx = fmaxf(mx, loc[i]);
    }
    red[t] = mx;
    __syncthreads();
    for (int s = 128; s > 0; s >>= 1) {
        if (t < s) red[t] = fmaxf(red[t], red[t + s]);
        __syncthreads();
    }
    mx = red[0];
    __syncthreads();

    float sum = 0.0f;
    #pragma unroll
    for (int i = 0; i < 8; ++i) {
        loc[i] = __expf(loc[i] - mx);
        sum += loc[i];
    }
    red[t] = sum;
    __syncthreads();
    for (int s = 128; s > 0; s >>= 1) {
        if (t < s) red[t] += red[t + s];
        __syncthreads();
    }
    const float inv = 1.0f / red[0];

    float*  of = outf + (size_t)row * L;
    __bf16* ob = (__bf16*)outbf + (size_t)row * L;
    #pragma unroll
    for (int i = 0; i < 8; ++i) {
        const float v = loc[i] * inv;
        of[t + i * 256] = v;
        ob[t + i * 256] = (__bf16)v;
    }
}

extern "C" void kernel_launch(void* const* d_in, const int* in_sizes, int n_in,
                              void* d_out, int out_size, void* d_ws, size_t ws_size,
                              hipStream_t stream)
{
    const int Bn = 8, SQ = 2048, SK = 2048, D = 1024;
    const float scale = 0.03125f; // 1/sqrt(1024)

    // d_in: 0=Q 1=K 2=V 3=attn_mask 4=WQ_w 5=WQ_b 6=WK_w 7=WK_b 8=WV_w 9=WV_b
    char* ws = (char*)d_ws;
    void*  qb = (void*)(ws);                         // bf16 [B*SQ, D]      33.5 MB
    void*  kb = (void*)(ws + (size_t)33554432);      // bf16 [B*SK, D]      33.5 MB
    void*  vT = (void*)(ws + (size_t)67108864);      // bf16 [B][D][SQ]     33.5 MB
    float* S  = (float*)(ws + (size_t)100663296);    // f32  [B][SK][SQ]   134.2 MB
    void*  ab = (void*)(ws + (size_t)234881024);     // bf16 [B][SK][SQ]    67.1 MB

    float* ctx  = (float*)d_out;                       // [B][SK][D]
    float* attn = (float*)d_out + (size_t)Bn * SK * D; // [B][SK][SQ]

    dim3 blk(256);

    // Pass 1: projections (fp32 A and W, bf16 out)
    gemm_wmma_k<0, true, true><<<dim3(D / 128, (Bn * SQ) / 128, 1), blk, 0, stream>>>(
        d_in[0], d_in[4], qb, (const float*)d_in[5], nullptr,
        Bn * SQ, D, D, 0, 0, 0, 0, SQ, 0.0f);
    gemm_wmma_k<0, true, true><<<dim3(D / 128, (Bn * SK) / 128, 1), blk, 0, stream>>>(
        d_in[1], d_in[6], kb, (const float*)d_in[7], nullptr,
        Bn * SK, D, D, 0, 0, 0, 0, SK, 0.0f);
    gemm_wmma_k<1, true, true><<<dim3(D / 128, (Bn * SQ) / 128, 1), blk, 0, stream>>>(
        d_in[2], d_in[8], vT, (const float*)d_in[9], nullptr,
        Bn * SQ, D, D, 0, 0, 0, 0, SQ, 0.0f);

    // Pass 2: S = (k . q^T)*scale + mask^T * (-1e9), batched over B
    gemm_wmma_k<2, false, false><<<dim3(SQ / 128, SK / 128, Bn), blk, 0, stream>>>(
        kb, qb, S, nullptr, (const int*)d_in[3],
        SK, SQ, D,
        (long long)SK * D, (long long)SQ * D,
        (long long)SK * SQ, (long long)SQ * SK,
        SK /* mask leading dim */, scale);

    // Pass 3: softmax over SQ per (b, k) row
    softmax_rows<<<dim3(Bn * SK), blk, 0, stream>>>(S, attn, ab, SQ);

    // Pass 4: context = attn_bf16 . v^T, batched over B
    gemm_wmma_k<3, false, false><<<dim3(D / 128, SK / 128, Bn), blk, 0, stream>>>(
        ab, vT, ctx, nullptr, nullptr,
        SK, D, SQ,
        (long long)SK * SQ, (long long)D * SQ,
        (long long)SK * D, 0, SQ, 1.0f);
}
